// LuongAttention_3427383902712
// MI455X (gfx1250) — matrix operand order repaired and checked by reading, not compiled, not run on hardware
//
#include <hip/hip_runtime.h>

// ---------------------------------------------------------------------------
// Luong attention for MI455X (gfx1250, wave32, WMMA).
//   out = softmax( (query@W^T + b) @ mem^T + (1-mask)*NEG ) @ mem
// All three GEMMs use v_wmma_f32_16x16x32_f16 with 2x4 register blocking:
// each wave owns a 32x64 C macro-tile (8 WMMA accumulators) -> 0.75 fragment
// loads per WMMA.  The 4 waves of a block stack along M (block tile 128x64),
// so all waves share identical B fragments (served once via WGP$/L2).
// Workspace layout (~194 MiB):
//   [0)       memT : B*D*Sk f16  (32 MiB)  transposed memories (for out GEMM)
//   [+32MiB)  wH   : D*D f16     ( 2 MiB)  W[o][d], N x K layout
//   [+34MiB)  qpH  : B*Sq*D f16  (32 MiB)  projected queries
//   [+66MiB)  S    : B*Sq*Sk f32 (128 MiB) logits -> softmax in place
// ---------------------------------------------------------------------------

typedef __attribute__((ext_vector_type(16))) _Float16 v16h;
typedef __attribute__((ext_vector_type(8)))  _Float16 v8h;
typedef __attribute__((ext_vector_type(4)))  _Float16 v4h;
typedef __attribute__((ext_vector_type(8)))  float    v8f;

constexpr int Bb = 8, Sq = 2048, Sk = 2048, Dd = 1024;
constexpr float NEG_BIG = -1000000000.0f;

#define HAS_ASYNC_LDS                                            \
  (__has_builtin(__builtin_amdgcn_global_load_async_to_lds_b32) && \
   __has_builtin(__builtin_amdgcn_s_wait_asynccnt))

#if HAS_ASYNC_LDS
typedef __attribute__((address_space(1))) int as1_i32;   // global
typedef __attribute__((address_space(3))) int as3_i32;   // LDS
#endif

// ---------------------------------------------------------------------------
// fp32 -> fp16 convert, 4 elements per thread (used for W)
__global__ __launch_bounds__(256)
void cvt_f32_to_f16_kernel(const float* __restrict__ src,
                           _Float16* __restrict__ dst, int n4) {
  int i = blockIdx.x * 256 + threadIdx.x;
  if (i >= n4) return;
  float4 f = ((const float4*)src)[i];
  v4h h;
  h[0] = (_Float16)f.x; h[1] = (_Float16)f.y;
  h[2] = (_Float16)f.z; h[3] = (_Float16)f.w;
  ((v4h*)dst)[i] = h;
}

// ---------------------------------------------------------------------------
// LDS-tiled transpose + f32->f16: memT[b][d][k] = (f16) mem[b][k][d]
// 32x32 tiles; global->LDS stage uses CDNA5 async-to-LDS when available.
__global__ __launch_bounds__(256)
void transpose_cvt_kernel(const float* __restrict__ Mem,
                          _Float16* __restrict__ MemT) {
  __shared__ float tile[32][33];
  const int b  = blockIdx.z;
  const int k0 = blockIdx.x * 32;   // sequence dim
  const int d0 = blockIdx.y * 32;   // feature dim
  const int tx = threadIdx.x;       // 0..31
  const int ty = threadIdx.y;       // 0..7
  const float* src = Mem + ((size_t)b * Sk + k0) * Dd + d0;
#if HAS_ASYNC_LDS
#pragma unroll
  for (int r = ty; r < 32; r += 8)
    __builtin_amdgcn_global_load_async_to_lds_b32(
        (as1_i32*)(src + (size_t)r * Dd + tx),
        (as3_i32*)&tile[r][tx],
        0, 0);
  __builtin_amdgcn_s_wait_asynccnt(0);
  __syncthreads();
#else
#pragma unroll
  for (int r = ty; r < 32; r += 8)
    tile[r][tx] = src[(size_t)r * Dd + tx];
  __syncthreads();
#endif
  _Float16* dst = MemT + ((size_t)b * Dd + d0) * Sk + k0;
#pragma unroll
  for (int r = ty; r < 32; r += 8)
    dst[(size_t)r * Sk + tx] = (_Float16)tile[tx][r];
}

// ---------------------------------------------------------------------------
// Fragment helpers (16-bit WMMA operand layouts, wave32):
//  A (16x32, MxK): row = lane&15; lanes<16 hold K kb..kb+15,
//                  lanes>=16 hold K kb+16..kb+31.
//  B (32x16, KxN): col = lane&15; lanes<16 hold K kb..kb+15,
//                  lanes>=16 hold K kb+16..kb+31.
__device__ inline v16h load_a_f16(const _Float16* __restrict__ arow, int kA0) {
  v8h lo = *(const v8h*)(arow + kA0);
  v8h hi = *(const v8h*)(arow + kA0 + 8);
  v16h a;
#pragma unroll
  for (int e = 0; e < 8; ++e) { a[e] = lo[e]; a[e + 8] = hi[e]; }
  return a;
}

__device__ inline v16h load_a_f32_cvt(const float* __restrict__ arow, int kA0) {
  float4 f0 = *(const float4*)(arow + kA0);
  float4 f1 = *(const float4*)(arow + kA0 + 4);
  float4 f2 = *(const float4*)(arow + kA0 + 8);
  float4 f3 = *(const float4*)(arow + kA0 + 12);
  v16h a;
  a[0] = (_Float16)f0.x;  a[1] = (_Float16)f0.y;
  a[2] = (_Float16)f0.z;  a[3] = (_Float16)f0.w;
  a[4] = (_Float16)f1.x;  a[5] = (_Float16)f1.y;
  a[6] = (_Float16)f1.z;  a[7] = (_Float16)f1.w;
  a[8] = (_Float16)f2.x;  a[9] = (_Float16)f2.y;
  a[10] = (_Float16)f2.z; a[11] = (_Float16)f2.w;
  a[12] = (_Float16)f3.x; a[13] = (_Float16)f3.y;
  a[14] = (_Float16)f3.z; a[15] = (_Float16)f3.w;
  return a;
}

__device__ inline v16h load_b_nk(const _Float16* __restrict__ brow, int kB0) {
  v8h b0 = *(const v8h*)(brow + kB0);
  v8h b1 = *(const v8h*)(brow + kB0 + 8);
  v16h b;
#pragma unroll
  for (int e = 0; e < 8; ++e) { b[e] = b0[e]; b[e + 8] = b1[e]; }
  return b;
}

__device__ inline v16h load_b_nk_f32(const float* __restrict__ brow, int kB0) {
  float4 f0 = *(const float4*)(brow + kB0);
  float4 f1 = *(const float4*)(brow + kB0 + 4);
  float4 f2 = *(const float4*)(brow + kB0 + 8);
  float4 f3 = *(const float4*)(brow + kB0 + 12);
  v16h b;
  b[0] = (_Float16)f0.x;  b[1] = (_Float16)f0.y;
  b[2] = (_Float16)f0.z;  b[3] = (_Float16)f0.w;
  b[4] = (_Float16)f1.x;  b[5] = (_Float16)f1.y;
  b[6] = (_Float16)f1.z;  b[7] = (_Float16)f1.w;
  b[8] = (_Float16)f2.x;  b[9] = (_Float16)f2.y;
  b[10] = (_Float16)f2.z; b[11] = (_Float16)f2.w;
  b[12] = (_Float16)f3.x; b[13] = (_Float16)f3.y;
  b[14] = (_Float16)f3.z; b[15] = (_Float16)f3.w;
  return b;
}

// ---------------------------------------------------------------------------
// Projection: qproj[m,n] = f16( sum_k query[m,k]*Wh[n,k] + bias[n] )
// M = B*Sq = 16384, N = K = D = 1024.  Block tile 128x64, wave tile 32x64.
__global__ __launch_bounds__(128)
void proj_gemm_kernel(const float* __restrict__ Q,
                      const _Float16* __restrict__ Wh,
                      const float* __restrict__ bias,
                      _Float16* __restrict__ Out) {
  const int N = Dd, K = Dd;
  const int wave = threadIdx.x >> 5;
  const int lane = threadIdx.x & 31;
  const bool hi = lane >= 16;
  const int l15 = lane & 15;
  const int tilesN = N / 64;                       // 16
  const int gm = blockIdx.x / tilesN;              // 0..127
  const int tn = blockIdx.x % tilesN;
  const int tm = gm * 4 + wave;                    // waves stack along M

  const float* arow[2];
  const _Float16* brow[4];
#pragma unroll
  for (int i = 0; i < 2; ++i)
    arow[i] = Q + (size_t)(tm * 32 + i * 16 + l15) * K;
#pragma unroll
  for (int j = 0; j < 4; ++j)
    brow[j] = Wh + (size_t)(tn * 64 + j * 16 + l15) * K;

  v8f acc[2][4] = {};
  for (int k = 0; k < K; k += 32) {
    const int kA0 = k + (hi ? 16 : 0);
    const int kB0 = k + (hi ? 16 : 0);
    __builtin_prefetch(arow[0] + k + 32, 0, 1);
    __builtin_prefetch(arow[1] + k + 32, 0, 1);
    v16h a[2], bm[4];
#pragma unroll
    for (int i = 0; i < 2; ++i) a[i] = load_a_f32_cvt(arow[i], kA0);
#pragma unroll
    for (int j = 0; j < 4; ++j) bm[j] = load_b_nk(brow[j], kB0);
#pragma unroll
    for (int i = 0; i < 2; ++i)
#pragma unroll
      for (int j = 0; j < 4; ++j)
        acc[i][j] = __builtin_amdgcn_wmma_f32_16x16x32_f16(
            false, a[i], false, bm[j], (short)0, acc[i][j], false, false);
  }
  float bv[4];
#pragma unroll
  for (int j = 0; j < 4; ++j) bv[j] = bias[tn * 64 + j * 16 + l15];
#pragma unroll
  for (int i = 0; i < 2; ++i)
#pragma unroll
    for (int r = 0; r < 8; ++r) {
      const int orow = tm * 32 + i * 16 + r + (hi ? 8 : 0);
#pragma unroll
      for (int j = 0; j < 4; ++j)
        Out[(size_t)orow * N + tn * 64 + j * 16 + l15] =
            (_Float16)(acc[i][j][r] + bv[j]);
    }
}

// ---------------------------------------------------------------------------
// Logits: S[b,q,k] = sum_d qp[b,q,d]*mem[b,k,d] + (1-mask)*NEG
// A = qproj f16, B = memories f32 rows (converted in-register).
__global__ __launch_bounds__(128)
void logits_gemm_kernel(const _Float16* __restrict__ Qp,
                        const float* __restrict__ MemF,
                        const int* __restrict__ mask,
                        float* __restrict__ S) {
  const int wave = threadIdx.x >> 5;
  const int lane = threadIdx.x & 31;
  const bool hi = lane >= 16;
  const int l15 = lane & 15;
  const int tilesN = Sk / 64;                      // 32
  const int groupsPerBatch = (Sq / 128) * tilesN;  // 16*32 = 512
  const int b = blockIdx.x / groupsPerBatch;
  const int g = blockIdx.x % groupsPerBatch;
  const int gm = g / tilesN, tn = g % tilesN;
  const int tm = gm * 4 + wave;                    // waves stack along M

  const _Float16* arow[2];
  const float* brow[4];
#pragma unroll
  for (int i = 0; i < 2; ++i)
    arow[i] = Qp + ((size_t)b * Sq + tm * 32 + i * 16 + l15) * Dd;
#pragma unroll
  for (int j = 0; j < 4; ++j)
    brow[j] = MemF + ((size_t)b * Sk + tn * 64 + j * 16 + l15) * Dd;

  v8f acc[2][4] = {};
  for (int k = 0; k < Dd; k += 32) {
    const int kA0 = k + (hi ? 16 : 0);
    const int kB0 = k + (hi ? 16 : 0);
    __builtin_prefetch(arow[0] + k + 32, 0, 1);
    __builtin_prefetch(arow[1] + k + 32, 0, 1);
    v16h a[2], bm[4];
#pragma unroll
    for (int i = 0; i < 2; ++i) a[i] = load_a_f16(arow[i], kA0);
#pragma unroll
    for (int j = 0; j < 4; ++j) bm[j] = load_b_nk_f32(brow[j], kB0);
#pragma unroll
    for (int i = 0; i < 2; ++i)
#pragma unroll
      for (int j = 0; j < 4; ++j)
        acc[i][j] = __builtin_amdgcn_wmma_f32_16x16x32_f16(
            false, a[i], false, bm[j], (short)0, acc[i][j], false, false);
  }
#pragma unroll
  for (int i = 0; i < 2; ++i)
#pragma unroll
    for (int r = 0; r < 8; ++r) {
      const int orow = tm * 32 + i * 16 + r + (hi ? 8 : 0);
      const size_t rbase = ((size_t)b * Sq + orow) * (size_t)Sk;
#pragma unroll
      for (int j = 0; j < 4; ++j) {
        const size_t idx = rbase + tn * 64 + j * 16 + l15;
        S[idx] = acc[i][j][r] + (1.0f - (float)mask[idx]) * NEG_BIG;
      }
    }
}

// ---------------------------------------------------------------------------
// Row softmax over Sk, in place.  One 256-thread block per row.
__global__ __launch_bounds__(256)
void softmax_rows_kernel(float* __restrict__ S) {
  __shared__ float red[256];
  const int tid = threadIdx.x;
  float* p = S + (size_t)blockIdx.x * Sk;

  float m = -3.402823466e38f;
  for (int i = tid; i < Sk; i += 256) m = fmaxf(m, p[i]);
  red[tid] = m;
  __syncthreads();
  for (int s = 128; s > 0; s >>= 1) {
    if (tid < s) red[tid] = fmaxf(red[tid], red[tid + s]);
    __syncthreads();
  }
  const float mx = red[0];
  __syncthreads();

  float sum = 0.0f;
  for (int i = tid; i < Sk; i += 256) {
    float e = __expf(p[i] - mx);
    p[i] = e;
    sum += e;
  }
  red[tid] = sum;
  __syncthreads();
  for (int s = 128; s > 0; s >>= 1) {
    if (tid < s) red[tid] += red[tid + s];
    __syncthreads();
  }
  const float inv = 1.0f / red[0];
  __syncthreads();
  for (int i = tid; i < Sk; i += 256) p[i] *= inv;
}

// ---------------------------------------------------------------------------
// Output: out[b,q,d] = sum_k P[b,q,k]*mem[b,k,d]
// A = P f32 (converted in-register), B = memT f16 (N x K rows, contiguous).
__global__ __launch_bounds__(128)
void out_gemm_kernel(const float* __restrict__ P,
                     const _Float16* __restrict__ MemT,
                     float* __restrict__ Out) {
  const int wave = threadIdx.x >> 5;
  const int lane = threadIdx.x & 31;
  const bool hi = lane >= 16;
  const int l15 = lane & 15;
  const int tilesN = Dd / 64;                      // 16
  const int groupsPerBatch = (Sq / 128) * tilesN;  // 16*16 = 256
  const int b = blockIdx.x / groupsPerBatch;
  const int g = blockIdx.x % groupsPerBatch;
  const int gm = g / tilesN, tn = g % tilesN;
  const int tm = gm * 4 + wave;                    // waves stack along M

  const float* arow[2];
  const _Float16* brow[4];
#pragma unroll
  for (int i = 0; i < 2; ++i)
    arow[i] = P + ((size_t)b * Sq + tm * 32 + i * 16 + l15) * (size_t)Sk;
#pragma unroll
  for (int j = 0; j < 4; ++j)
    brow[j] = MemT + ((size_t)b * Dd + tn * 64 + j * 16 + l15) * (size_t)Sk;

  v8f acc[2][4] = {};
  for (int k = 0; k < Sk; k += 32) {
    const int kA0 = k + (hi ? 16 : 0);
    const int kB0 = k + (hi ? 16 : 0);
    __builtin_prefetch(arow[0] + k + 32, 0, 1);
    __builtin_prefetch(arow[1] + k + 32, 0, 1);
    v16h a[2], bm[4];
#pragma unroll
    for (int i = 0; i < 2; ++i) a[i] = load_a_f32_cvt(arow[i], kA0);
#pragma unroll
    for (int j = 0; j < 4; ++j) bm[j] = load_b_nk(brow[j], kB0);
#pragma unroll
    for (int i = 0; i < 2; ++i)
#pragma unroll
      for (int j = 0; j < 4; ++j)
        acc[i][j] = __builtin_amdgcn_wmma_f32_16x16x32_f16(
            false, a[i], false, bm[j], (short)0, acc[i][j], false, false);
  }
#pragma unroll
  for (int i = 0; i < 2; ++i)
#pragma unroll
    for (int r = 0; r < 8; ++r) {
      const int orow = tm * 32 + i * 16 + r + (hi ? 8 : 0);
#pragma unroll
      for (int j = 0; j < 4; ++j)
        Out[((size_t)b * Sq + orow) * Dd + tn * 64 + j * 16 + l15] =
            acc[i][j][r];
    }
}

// ---------------------------------------------------------------------------
extern "C" void kernel_launch(void* const* d_in, const int* in_sizes, int n_in,
                              void* d_out, int out_size, void* d_ws, size_t ws_size,
                              hipStream_t stream) {
  const float* query    = (const float*)d_in[0];   // [B,Sq,D] f32
  const float* memories = (const float*)d_in[1];   // [B,Sk,D] f32
  const int*   mask     = (const int*)d_in[2];     // [B,Sq,Sk] i32
  const float* W        = (const float*)d_in[3];   // [D,D] f32
  const float* bias     = (const float*)d_in[4];   // [D] f32
  float*       out      = (float*)d_out;           // [B,Sq,D] f32

  char* ws = (char*)d_ws;
  _Float16* memT = (_Float16*)ws;
  size_t off = (size_t)Bb * Dd * Sk * sizeof(_Float16);
  _Float16* wH = (_Float16*)(ws + off);
  off += (size_t)Dd * Dd * sizeof(_Float16);
  _Float16* qpH = (_Float16*)(ws + off);
  off += (size_t)Bb * Sq * Dd * sizeof(_Float16);
  float* Sbuf = (float*)(ws + off);

  // 1) transpose + convert memories: memT[b][d][k] = f16(mem[b][k][d])
  transpose_cvt_kernel<<<dim3(Sk / 32, Dd / 32, Bb), dim3(32, 8), 0, stream>>>(
      memories, memT);

  // 2) convert W to f16 (kept [o][d] = N x K)
  {
    int n4 = Dd * Dd / 4;                                         // 262,144
    cvt_f32_to_f16_kernel<<<n4 / 256, 256, 0, stream>>>(W, wH, n4);
  }

  // 3) projection GEMM: (16384/128)*(1024/64) = 2048 blocks
  proj_gemm_kernel<<<2048, 128, 0, stream>>>(query, wH, bias, qpH);

  // 4) logits GEMM + mask: 8*(2048/128)*(2048/64) = 4096 blocks
  logits_gemm_kernel<<<4096, 128, 0, stream>>>(qpH, memories, mask, Sbuf);

  // 5) softmax: one block per row, B*Sq = 16384 rows
  softmax_rows_kernel<<<Bb * Sq, 256, 0, stream>>>(Sbuf);

  // 6) output GEMM: 8*(2048/128)*(1024/64) = 2048 blocks
  out_gemm_kernel<<<2048, 128, 0, stream>>>(Sbuf, memT, out);
}
